// CaNetConv_2602750181782
// MI455X (gfx1250) — matrix operand order repaired and checked by reading, not compiled, not run on hardware
//
#include <hip/hip_runtime.h>
#include <hip/hip_bf16.h>

typedef __attribute__((ext_vector_type(16))) __bf16         v16bf;
typedef __attribute__((ext_vector_type(8)))  float          v8f;
typedef __attribute__((ext_vector_type(8)))  unsigned short u16x8;
typedef __attribute__((ext_vector_type(16))) unsigned short u16x16;

struct U16Pair { u16x8 lo; u16x8 hi; };   // 32 bytes -> v16bf

static constexpr int       N_   = 50000;
static constexpr int       F_   = 128;
static constexpr int       K_   = 8;
static constexpr long long E_   = 1600000;
static constexpr int       TWOF = 256;

// workspace layout (bytes), all 512-aligned
static constexpr size_t DEG_OFF  = 0;                       // N floats
static constexpr size_t HI_OFF   = 200704;                  // N*F floats
static constexpr size_t HI_BYTES = (size_t)N_ * F_ * 4;     // 25,600,000
static constexpr size_t A_OFF    = HI_OFF + HI_BYTES;       // N*2F bf16
static constexpr size_t A_BYTES  = (size_t)N_ * TWOF * 2;   // 25,600,000
static constexpr size_t WB_OFF   = A_OFF + A_BYTES;         // K*2F*F bf16 = 512KB

__device__ __forceinline__ unsigned short f2bf(float f) {
    unsigned u = __builtin_bit_cast(unsigned, f);
    unsigned r = u + 0x7FFFu + ((u >> 16) & 1u);   // round-to-nearest-even
    return (unsigned short)(r >> 16);
}

// ---- 1) degree[col] += 1 ------------------------------------------------
__global__ void degree_kernel(const long long* __restrict__ adj, float* __restrict__ deg) {
    long long e = (long long)blockIdx.x * blockDim.x + threadIdx.x;
    if (e < E_) atomicAdd(&deg[(int)adj[E_ + e]], 1.0f);
}

// ---- 2) deg -> 1/sqrt(deg) (0 if deg==0, matches nan_to_num) ------------
__global__ void invsqrt_kernel(float* __restrict__ deg) {
    int n = blockIdx.x * blockDim.x + threadIdx.x;
    if (n < N_) {
        float d = deg[n];
        deg[n] = (d > 0.0f) ? rsqrtf(d) : 0.0f;
    }
}

// ---- 3) hi[col] += val * x[row]   (one wave32 per edge, float4/lane) ----
__global__ void scatter_kernel(const long long* __restrict__ adj,
                               const float* __restrict__ x,
                               const float* __restrict__ invs,
                               float* __restrict__ hi) {
    int lane = threadIdx.x & 31;
    long long e = (long long)blockIdx.x * (blockDim.x >> 5) + (threadIdx.x >> 5);
    if (e >= E_) return;
    int r = (int)adj[e];
    int c = (int)adj[E_ + e];
    float val = invs[r] * invs[c];
    float4 xv = ((const float4*)x)[(long long)r * 32 + lane];
    float* dst = hi + (long long)c * F_ + lane * 4;
    atomicAdd(dst + 0, val * xv.x);
    atomicAdd(dst + 1, val * xv.y);
    atomicAdd(dst + 2, val * xv.z);
    atomicAdd(dst + 3, val * xv.w);
}

// ---- 4) A[n, 0:128]=bf16(hi), A[n,128:256]=bf16(x) ----------------------
__global__ void packA_kernel(const float* __restrict__ hi,
                             const float* __restrict__ x,
                             unsigned short* __restrict__ A) {
    long long i = (long long)blockIdx.x * blockDim.x + threadIdx.x;
    if (i >= (long long)N_ * TWOF) return;
    int f = (int)(i & 255);
    long long n = i >> 8;
    float v = (f < F_) ? hi[n * F_ + f] : x[n * F_ + (f - F_)];
    A[i] = f2bf(v);
}

// ---- 5) pack weights into WMMA B-fragment order -------------------------
// Wb[k][otile][c][lane][j], j=0..15 : K = c*32 + 16*(j/8) + 8*(lane/16) + j%8,
//                                     Ncol = otile*16 + (lane%16)
__global__ void packW_kernel(const float* __restrict__ w, unsigned short* __restrict__ Wb) {
    int i = blockIdx.x * blockDim.x + threadIdx.x;     // < 262144
    if (i >= K_ * TWOF * F_) return;
    int j    =  i        & 15;
    int lane = (i >> 4)  & 31;
    int c    = (i >> 9)  & 7;
    int ot   = (i >> 12) & 7;
    int k    =  i >> 15;
    int f = c * 32 + ((j >> 3) << 4) + ((lane >> 4) << 3) + (j & 7);
    int o = ot * 16 + (lane & 15);
    Wb[i] = f2bf(w[((size_t)k * TWOF + f) * F_ + o]);
}

// ---- 6) gated expert GEMM + residual ------------------------------------
// block = 16 nodes, 8 waves; wave w computes expert k=w with 64 WMMAs,
// gates by e[n,k], reduces across experts via LDS atomics, adds x.
__global__ __launch_bounds__(256) void gemm_gate_kernel(
        const unsigned short* __restrict__ A,
        const unsigned short* __restrict__ Wb,
        const float* __restrict__ egate,
        const float* __restrict__ x,
        float* __restrict__ out) {
    __shared__ float s_acc[16 * 128];
    const int tid  = threadIdx.x;
    const int k    = tid >> 5;        // wave id == expert id
    const int lane = tid & 31;
    const int g    = lane >> 4;       // half-wave group
    const int nl   = lane & 15;
    const int nodeBase = blockIdx.x * 16;   // N divisible by 16

    #pragma unroll
    for (int t = 0; t < 8; ++t) s_acc[tid + t * 256] = 0.0f;
    __syncthreads();

    // A fragments: lane holds row m=nl; chunk c covers K=[32c,32c+32);
    // elements j<8 at K=32c+8g+j, j>=8 at K=32c+16+8g+(j-8)
    v16bf afrag[8];
    const unsigned short* arow = A + (size_t)(nodeBase + nl) * TWOF;
    #pragma unroll
    for (int c = 0; c < 8; ++c) {
        u16x8 lo = *(const u16x8*)(arow + c * 32 + 8 * g);
        u16x8 hi = *(const u16x8*)(arow + c * 32 + 16 + 8 * g);
        U16Pair p{lo, hi};
        afrag[c] = __builtin_bit_cast(v16bf, p);
    }

    // gate scalars for C rows m = v + 8g
    float ev[8];
    #pragma unroll
    for (int v = 0; v < 8; ++v)
        ev[v] = egate[(size_t)(nodeBase + v + 8 * g) * K_ + k];

    const unsigned short* wbase = Wb + (size_t)k * (8 * 8 * 32 * 16);
    #pragma unroll
    for (int ot = 0; ot < 8; ++ot) {
        v8f acc = {};
        #pragma unroll
        for (int c = 0; c < 8; ++c) {
            u16x16 braw = *(const u16x16*)(wbase + ((size_t)(ot * 8 + c) * 32 + lane) * 16);
            v16bf bfrag = __builtin_bit_cast(v16bf, braw);
            acc = __builtin_amdgcn_wmma_f32_16x16x32_bf16(
                false, afrag[c], false, bfrag, (short)0, acc, false, false);
        }
        // C layout: VGPR v, lane -> row m = v + 8g, col = nl
        const int o = ot * 16 + nl;
        #pragma unroll
        for (int v = 0; v < 8; ++v)
            atomicAdd(&s_acc[(v + 8 * g) * 128 + o], acc[v] * ev[v]);
    }
    __syncthreads();

    const float* xrow = x   + (size_t)nodeBase * F_;
    float*       orow = out + (size_t)nodeBase * F_;
    #pragma unroll
    for (int t = 0; t < 8; ++t) {
        int idx = tid + t * 256;
        orow[idx] = s_acc[idx] + xrow[idx];
    }
}

extern "C" void kernel_launch(void* const* d_in, const int* in_sizes, int n_in,
                              void* d_out, int out_size, void* d_ws, size_t ws_size,
                              hipStream_t stream) {
    const float*     x   = (const float*)d_in[0];
    const long long* adj = (const long long*)d_in[1];   // int64 [2,E]
    const float*     e   = (const float*)d_in[2];
    const float*     w   = (const float*)d_in[3];
    float*           out = (float*)d_out;
    char*            ws  = (char*)d_ws;

    float*          deg = (float*)(ws + DEG_OFF);
    float*          hi  = (float*)(ws + HI_OFF);
    unsigned short* A   = (unsigned short*)(ws + A_OFF);
    unsigned short* Wb  = (unsigned short*)(ws + WB_OFF);

    // zero deg + hi (fresh every call; ws is not re-poisoned between replays)
    hipMemsetAsync(ws, 0, HI_OFF + HI_BYTES, stream);

    degree_kernel <<<(unsigned)((E_ + 255) / 256), 256, 0, stream>>>(adj, deg);
    invsqrt_kernel<<<(N_ + 255) / 256,            256, 0, stream>>>(deg);
    scatter_kernel<<<(unsigned)(E_ / 8),          256, 0, stream>>>(adj, x, deg, hi);
    packA_kernel  <<<(unsigned)(((long long)N_ * TWOF + 255) / 256), 256, 0, stream>>>(hi, x, A);
    packW_kernel  <<<(K_ * TWOF * F_) / 256,      256, 0, stream>>>(w, Wb);
    gemm_gate_kernel<<<N_ / 16,                   256, 0, stream>>>(A, Wb, e, x, out);
}
